// MobiuAttention_8564164788652
// MI455X (gfx1250) — matrix operand order repaired and checked by reading, not compiled, not run on hardware
//
#include <hip/hip_runtime.h>
#include <hip/hip_bf16.h>
#include <math.h>

// ---------------------------------------------------------------------------
// MobiuAttention for MI455X (gfx1250, wave32).
//  * bf16x3 split-precision GEMMs on V_WMMA_F32_16X16X32_BF16, with X/W
//    pre-packed into bf16 hi/lo planes (conversion done once, not per tile).
//  * Each wave computes a 16x64 strip (A-fragment reuse across 4 N-subtiles).
//  * Sequential decay scan: 1 block per (b,h) chain, 64x64 state in registers;
//    q/k/v broadcast into LDS per step via GLOBAL_LOAD_ASYNC_TO_LDS_B32
//    (ASYNCcnt) -- the CDNA5 async global->LDS path.
// ---------------------------------------------------------------------------

typedef __attribute__((ext_vector_type(16))) __bf16 v16bf;
typedef __attribute__((ext_vector_type(8)))  __bf16 v8bf;
typedef __attribute__((ext_vector_type(4)))  __bf16 v4bf;
typedef __attribute__((ext_vector_type(8)))  float  v8f;

#define BB 4
#define TT 2048
#define DD 1024
#define HH 16
#define EE 64
#define DHH 256
#define MM (BB*TT)   // 8192 rows

static __device__ __forceinline__ v16bf frag16(const __bf16* p0, const __bf16* p1) {
    v8bf x = *(const v8bf*)p0;
    v8bf y = *(const v8bf*)p1;
    return __builtin_shufflevector(x, y, 0,1,2,3,4,5,6,7,8,9,10,11,12,13,14,15);
}

// ---------------------------------------------------------------------------
// Pack f32 -> bf16 hi/lo planes (done once per tensor; bandwidth-trivial).
// ---------------------------------------------------------------------------
__global__ __launch_bounds__(256)
void pack_bf16(const float* __restrict__ in, __bf16* __restrict__ hi,
               __bf16* __restrict__ lo, long n)
{
    long i = ((long)blockIdx.x * 256 + threadIdx.x) * 4;
    if (i >= n) return;
    float4 f = *(const float4*)(in + i);
    v4bf h, l;
    float fv[4] = {f.x, f.y, f.z, f.w};
    #pragma unroll
    for (int j = 0; j < 4; ++j) {
        __bf16 hh = (__bf16)fv[j];
        h[j] = hh;
        l[j] = (__bf16)(fv[j] - (float)hh);
    }
    *(v4bf*)(hi + i) = h;
    *(v4bf*)(lo + i) = l;
}

// ---------------------------------------------------------------------------
// GEMM: Y[M,N] = act( X[M,K] @ W[N,K]^T + bias ), X/W given as bf16 hi/lo
// planes. One wave computes 16 x (16*NSUB); 3 WMMAs per subtile per k-step.
// ACT: 0=identity 1=tanh 2=sigmoid.  OUT: 0=f32, 1=bf16 hi/lo planes.
// ---------------------------------------------------------------------------
template<int ACT, int NSUB, int OUT>
__global__ __launch_bounds__(256)
void gemm_bf16x3(const __bf16* __restrict__ Xh, const __bf16* __restrict__ Xl,
                 const __bf16* __restrict__ Wh, const __bf16* __restrict__ Wl,
                 const float*  __restrict__ bias,
                 float* __restrict__ Yf, __bf16* __restrict__ Yh, __bf16* __restrict__ Yl,
                 int M, int N, int K)
{
    const int wave = threadIdx.x >> 5;
    const int lane = threadIdx.x & 31;
    const int nTilesN = N / (16 * NSUB);
    const int nTiles  = (M >> 4) * nTilesN;
    const int tile = blockIdx.x * 8 + wave;
    if (tile >= nTiles) return;             // uniform per wave: EXEC stays all-1s

    const int tm   = tile / nTilesN;
    const int tn   = tile - tm * nTilesN;   // in units of 16*NSUB columns
    const int half = lane >> 4;
    const int l16  = lane & 15;

    // A fragment: lane holds row M=tm*16+l16; elems 0..7 at k0+half*8, 8..15 at +16
    const size_t aoff = (size_t)(tm * 16 + l16) * K + half * 8;
    // B fragment: lane holds col N; elems 0..7 at k0+half*8, 8..15 at +16
    const size_t boff = (size_t)(tn * 16 * NSUB + l16) * K + half * 8;

    v8f acc[NSUB];
    #pragma unroll
    for (int j = 0; j < NSUB; ++j) acc[j] = (v8f){};

    for (int k0 = 0; k0 < K; k0 += 32) {
        const __bf16* ahp = Xh + aoff + k0;
        const __bf16* alp = Xl + aoff + k0;
        __builtin_prefetch(ahp + 32, 0, 0);       // global_prefetch next k-step
        __builtin_prefetch(alp + 32, 0, 0);
        v16bf ah = frag16(ahp, ahp + 16);
        v16bf al = frag16(alp, alp + 16);
        #pragma unroll
        for (int j = 0; j < NSUB; ++j) {
            const __bf16* bhp = Wh + boff + (size_t)j * 16 * K + k0;
            const __bf16* blp = Wl + boff + (size_t)j * 16 * K + k0;
            v16bf bh = frag16(bhp, bhp + 16);
            v16bf bl = frag16(blp, blp + 16);
            // bf16x3: hi*hi + hi*lo + lo*hi (f32 accumulate; lo*lo negligible)
            acc[j] = __builtin_amdgcn_wmma_f32_16x16x32_bf16(false, ah, false, bh, (short)0, acc[j], false, false);
            acc[j] = __builtin_amdgcn_wmma_f32_16x16x32_bf16(false, ah, false, bl, (short)0, acc[j], false, false);
            acc[j] = __builtin_amdgcn_wmma_f32_16x16x32_bf16(false, al, false, bh, (short)0, acc[j], false, false);
        }
    }

    // C/D layout: VGPR r, lane -> M = tm*16 + half*8 + r, N = tile_n + l16
    #pragma unroll
    for (int j = 0; j < NSUB; ++j) {
        const int ncol = tn * 16 * NSUB + j * 16 + l16;
        const float bv = bias ? bias[ncol] : 0.0f;
        #pragma unroll
        for (int r = 0; r < 8; ++r) {
            const int mrow = tm * 16 + half * 8 + r;
            float v = acc[j][r] + bv;
            if (ACT == 1) v = tanhf(v);
            else if (ACT == 2) v = 1.0f / (1.0f + __expf(-v));
            const size_t o = (size_t)mrow * N + ncol;
            if (OUT == 0) {
                Yf[o] = v;
            } else {
                __bf16 hh = (__bf16)v;
                Yh[o] = hh;
                Yl[o] = (__bf16)(v - (float)hh);
            }
        }
    }
}

// ---------------------------------------------------------------------------
// Decay scan: S[e,f] = S[e,f]*lam[t,e] + k[t,e]*v[t,f]; out[t,f]=sum_e q[t,e]S[e,f]
// grid = B*H blocks of 256 threads; thread (eg,f): f = tid&63, owns e-rows
// eg*16..+15 of column f (16 state regs). Per-step q/k/v broadcast uses
// GLOBAL_LOAD_ASYNC_TO_LDS_B32 (waves 0-5, 1 dword/lane), ASYNCcnt-tracked.
// Output written directly as bf16 hi/lo planes for the final WMMA GEMM.
// ---------------------------------------------------------------------------
__global__ __launch_bounds__(256)
void mobiu_scan(const float* __restrict__ q, const float* __restrict__ k,
                const float* __restrict__ v, const float* __restrict__ lc,
                const float* __restrict__ decay,
                __bf16* __restrict__ yh, __bf16* __restrict__ yl)
{
    const int b   = blockIdx.x >> 4;     // H = 16
    const int h   = blockIdx.x & 15;
    const int tid = threadIdx.x;
    const int f   = tid & 63;
    const int eg  = tid >> 6;            // 0..3

    __shared__ float base_sh[EE];
    __shared__ float qsh[EE], ksh[EE], vsh[EE];
    __shared__ float red[4][EE];
    __shared__ float lc_sh;

    if (tid < EE) {
        const float dp = decay[h * EE + tid];
        base_sh[tid] = 1.0f / (1.0f + __expf(-dp));   // sigmoid(decay_params)
    }

    // async-copy roles: waves 0-1 -> q, 2-3 -> k, 4-5 -> v (full-EXEC waves)
    const float* gbase = nullptr;
    unsigned int ldsaddr = 0;
    if (tid < 64)        { gbase = q + tid;       ldsaddr = (unsigned int)(uintptr_t)&qsh[tid]; }
    else if (tid < 128)  { gbase = k + tid - 64;  ldsaddr = (unsigned int)(uintptr_t)&ksh[tid - 64]; }
    else if (tid < 192)  { gbase = v + tid - 128; ldsaddr = (unsigned int)(uintptr_t)&vsh[tid - 128]; }

    __syncthreads();

    float basereg[16], s[16];
    #pragma unroll
    for (int i = 0; i < 16; ++i) { basereg[i] = base_sh[eg * 16 + i]; s[i] = 0.0f; }

    for (int t = 0; t < TT; ++t) {
        const size_t idx = (((size_t)b * TT + t) * HH + h) * EE;  // into [B,T,H,E]
        if (tid < 192) {
            unsigned long long ga = (unsigned long long)(uintptr_t)(gbase + idx);
            asm volatile("global_load_async_to_lds_b32 %0, %1, off"
                         :: "v"(ldsaddr), "v"(ga) : "memory");
        }
        if (tid == 192) lc_sh = lc[((size_t)b * TT + t) * HH + h];
        asm volatile("s_wait_asynccnt 0x0" ::: "memory");
        __syncthreads();

        const float scale = 1.0f + 0.2f * lc_sh;
        const float vf = vsh[f];
        float partial = 0.0f;
        #pragma unroll
        for (int i = 0; i < 16; ++i) {
            const int e = eg * 16 + i;
            const float lam = fminf(basereg[i] * scale, 0.9995f);
            s[i] = s[i] * lam + ksh[e] * vf;
            partial += qsh[e] * s[i];
        }
        red[eg][f] = partial;
        __syncthreads();
        if (eg == 0) {
            const float r = red[0][f] + red[1][f] + red[2][f] + red[3][f];
            const __bf16 hh = (__bf16)r;
            yh[idx + f] = hh;
            yl[idx + f] = (__bf16)(r - (float)hh);
        }
        // qsh/ksh/vsh rewritten only after every thread passed the barrier
        // above; red rewritten only after the next barrier.
    }
}

// ---------------------------------------------------------------------------
extern "C" void kernel_launch(void* const* d_in, const int* in_sizes, int n_in,
                              void* d_out, int out_size, void* d_ws, size_t ws_size,
                              hipStream_t stream) {
    const float* x     = (const float*)d_in[0];
    const float* q_w   = (const float*)d_in[1];
    const float* k_w   = (const float*)d_in[2];
    const float* v_w   = (const float*)d_in[3];
    const float* o_w   = (const float*)d_in[4];
    const float* cs_w1 = (const float*)d_in[5];
    const float* cs_b1 = (const float*)d_in[6];
    const float* cs_w2 = (const float*)d_in[7];
    const float* cs_b2 = (const float*)d_in[8];
    const float* decay = (const float*)d_in[9];
    float* out = (float*)d_out;

    // ---- workspace layout (all chunks >=64B aligned) ----
    char* ws = (char*)d_ws;
    size_t off = 0;
    auto alloc = [&](size_t bytes) { void* p = ws + off; off += (bytes + 63) & ~(size_t)63; return p; };
    float*  qf  = (float*)alloc((size_t)MM * DD * 4);
    float*  kf  = (float*)alloc((size_t)MM * DD * 4);
    float*  vf  = (float*)alloc((size_t)MM * DD * 4);
    __bf16* xh  = (__bf16*)alloc((size_t)MM * DD * 2);
    __bf16* xl  = (__bf16*)alloc((size_t)MM * DD * 2);
    __bf16* yhp = (__bf16*)alloc((size_t)MM * DD * 2);
    __bf16* ylp = (__bf16*)alloc((size_t)MM * DD * 2);
    __bf16* qwh = (__bf16*)alloc((size_t)DD * DD * 2);
    __bf16* qwl = (__bf16*)alloc((size_t)DD * DD * 2);
    __bf16* kwh = (__bf16*)alloc((size_t)DD * DD * 2);
    __bf16* kwl = (__bf16*)alloc((size_t)DD * DD * 2);
    __bf16* vwh = (__bf16*)alloc((size_t)DD * DD * 2);
    __bf16* vwl = (__bf16*)alloc((size_t)DD * DD * 2);
    __bf16* owh = (__bf16*)alloc((size_t)DD * DD * 2);
    __bf16* owl = (__bf16*)alloc((size_t)DD * DD * 2);
    __bf16* w1h = (__bf16*)alloc((size_t)DHH * DD * 2);
    __bf16* w1l = (__bf16*)alloc((size_t)DHH * DD * 2);
    __bf16* w2h = (__bf16*)alloc((size_t)HH * DHH * 2);
    __bf16* w2l = (__bf16*)alloc((size_t)HH * DHH * 2);
    __bf16* h1h = (__bf16*)alloc((size_t)MM * DHH * 2);
    __bf16* h1l = (__bf16*)alloc((size_t)MM * DHH * 2);
    float*  lcf = (float*)alloc((size_t)MM * HH * 4);

    dim3 blk(256);
    auto packb = [](long n) { return (unsigned)((n / 4 + 255) / 256); };
    auto nblocks = [](int m, int n, int nsub) { return (((m >> 4) * (n / (16 * nsub))) + 7) / 8; };

    // ---- pack f32 -> bf16 hi/lo planes (once) ----
    pack_bf16<<<packb((long)MM * DD), blk, 0, stream>>>(x,     xh,  xl,  (long)MM * DD);
    pack_bf16<<<packb((long)DD * DD), blk, 0, stream>>>(q_w,   qwh, qwl, (long)DD * DD);
    pack_bf16<<<packb((long)DD * DD), blk, 0, stream>>>(k_w,   kwh, kwl, (long)DD * DD);
    pack_bf16<<<packb((long)DD * DD), blk, 0, stream>>>(v_w,   vwh, vwl, (long)DD * DD);
    pack_bf16<<<packb((long)DD * DD), blk, 0, stream>>>(o_w,   owh, owl, (long)DD * DD);
    pack_bf16<<<packb((long)DHH * DD), blk, 0, stream>>>(cs_w1, w1h, w1l, (long)DHH * DD);
    pack_bf16<<<packb((long)HH * DHH), blk, 0, stream>>>(cs_w2, w2h, w2l, (long)HH * DHH);

    // ---- projections: q/k/v = x @ W^T (f32 out for the recurrence) ----
    gemm_bf16x3<0,4,0><<<nblocks(MM, DD, 4), blk, 0, stream>>>(xh, xl, qwh, qwl, nullptr, qf, nullptr, nullptr, MM, DD, DD);
    gemm_bf16x3<0,4,0><<<nblocks(MM, DD, 4), blk, 0, stream>>>(xh, xl, kwh, kwl, nullptr, kf, nullptr, nullptr, MM, DD, DD);
    gemm_bf16x3<0,4,0><<<nblocks(MM, DD, 4), blk, 0, stream>>>(xh, xl, vwh, vwl, nullptr, vf, nullptr, nullptr, MM, DD, DD);
    // ---- sensor: h1 = tanh(x@w1^T+b1) (hi/lo out); lc = sigmoid(h1@w2^T+b2) ----
    gemm_bf16x3<1,4,1><<<nblocks(MM, DHH, 4), blk, 0, stream>>>(xh, xl, w1h, w1l, cs_b1, nullptr, h1h, h1l, MM, DHH, DD);
    gemm_bf16x3<2,1,0><<<nblocks(MM, HH, 1),  blk, 0, stream>>>(h1h, h1l, w2h, w2l, cs_b2, lcf, nullptr, nullptr, MM, HH, DHH);
    // ---- sequential decay scan (latency-bound core): 64 chains ----
    mobiu_scan<<<BB * HH, blk, 0, stream>>>(qf, kf, vf, lcf, decay, yhp, ylp);
    // ---- output projection ----
    gemm_bf16x3<0,4,0><<<nblocks(MM, DD, 4), blk, 0, stream>>>(yhp, ylp, owh, owl, nullptr, out, nullptr, nullptr, MM, DD, DD);
}
